// GraphConvolution_54975581388887
// MI455X (gfx1250) — matrix operand order repaired and checked by reading, not compiled, not run on hardware
//
#include <hip/hip_runtime.h>
#include <math.h>

// ---------------- problem constants (match reference) ----------------
#define N_NODES   100000
#define F_INPUT   64
#define E_EDGES   1000000
#define ES_PAIRS  1000000
#define G_GRAPHS  128

// ---------------- CDNA5 WMMA types ----------------
typedef __attribute__((ext_vector_type(16))) __bf16 v16bf;
typedef __attribute__((ext_vector_type(8)))  __bf16 v8bf;
typedef __attribute__((ext_vector_type(8)))  float  v8f;

__device__ __forceinline__ v8f wmma_bf16(v16bf a, v16bf b, v8f c) {
    // D = A(16x32 bf16) * B(32x16 bf16) + C(16x16 f32)
    return __builtin_amdgcn_wmma_f32_16x16x32_bf16(
        /*neg_a=*/false, a, /*neg_b=*/false, b,
        /*c_mod=*/(short)0, c, /*reuse_a=*/false, /*reuse_b=*/false);
}

// ---------------- utility kernels ----------------
__global__ void fill_kernel(float* __restrict__ p, float v, long n) {
    long i = (long)blockIdx.x * blockDim.x + threadIdx.x;
    if (i < n) p[i] = v;
}

__global__ void degree_kernel(const int* __restrict__ cols, const float* __restrict__ w,
                              float* __restrict__ deg, int E) {
    int e = blockIdx.x * blockDim.x + threadIdx.x;
    if (e < E) atomicAdd(&deg[cols[e]], w[e]);
}

__global__ void dinv_kernel(float* __restrict__ d, int n) {
    int i = blockIdx.x * blockDim.x + threadIdx.x;
    if (i < n) d[i] = rsqrtf(d[i] + 1.0f);   // +1 = self loop; always > 0
}

// ---------------- dense GEMM: C[M x NC] = A[M x K] @ B[K x NC] (+bias, relu) ----------------
// 128 threads = 4 waves; each wave computes a 16-row strip of all NC columns.
template<int K, int NC, int ACT>
__global__ __launch_bounds__(128)
void gemm_wmma(const float* __restrict__ A, const float* __restrict__ B,
               const float* __restrict__ bias, float* __restrict__ C, int M)
{
    __shared__ __bf16 Bs[NC][K];      // column-major: Bs[n][k]
    __shared__ float  biasS[NC];

    const int tid = threadIdx.x;
    for (int i = tid; i < K * NC; i += 128) {
        int k = i / NC, n = i % NC;
        Bs[n][k] = (__bf16)B[i];
    }
    if (tid < NC) biasS[tid] = bias ? bias[tid] : 0.0f;
    __syncthreads();

    const int wave = tid >> 5;
    const int lane = tid & 31;
    const int ln16 = lane & 15;
    const int hi   = lane >> 4;
    const int row0 = (blockIdx.x * 4 + wave) * 16;
    if (row0 >= M) return;            // wave-uniform (M multiple of 16)

    v8f acc[NC / 16];
#pragma unroll
    for (int n = 0; n < NC / 16; ++n) acc[n] = (v8f){0,0,0,0,0,0,0,0};

    const float* arow = A + (size_t)(row0 + ln16) * K;
#pragma unroll
    for (int k0 = 0; k0 < K; k0 += 32) {
        const int kb0 = k0 + hi * 16;
        const int kb1 = k0 + hi * 16 + 8;
        float4 p0 = *(const float4*)(arow + kb0);
        float4 p1 = *(const float4*)(arow + kb0 + 4);
        float4 p2 = *(const float4*)(arow + kb1);
        float4 p3 = *(const float4*)(arow + kb1 + 4);
        v16bf a;
        a[0]=(__bf16)p0.x; a[1]=(__bf16)p0.y; a[2]=(__bf16)p0.z; a[3]=(__bf16)p0.w;
        a[4]=(__bf16)p1.x; a[5]=(__bf16)p1.y; a[6]=(__bf16)p1.z; a[7]=(__bf16)p1.w;
        a[8]=(__bf16)p2.x; a[9]=(__bf16)p2.y; a[10]=(__bf16)p2.z; a[11]=(__bf16)p2.w;
        a[12]=(__bf16)p3.x; a[13]=(__bf16)p3.y; a[14]=(__bf16)p3.z; a[15]=(__bf16)p3.w;
#pragma unroll
        for (int n = 0; n < NC / 16; ++n) {
            v16bf b = *(const v16bf*)&Bs[n * 16 + ln16][k0 + hi * 16];
            acc[n] = wmma_bf16(a, b, acc[n]);
        }
    }

#pragma unroll
    for (int n = 0; n < NC / 16; ++n) {
#pragma unroll
        for (int r = 0; r < 8; ++r) {
            int m = row0 + hi * 8 + r;
            int c = n * 16 + ln16;
            float v = acc[n][r] + biasS[c];
            if (ACT) v = fmaxf(v, 0.0f);
            C[(size_t)m * NC + c] = v;
        }
    }
}

// ---------------- edge aggregation: agg[col] += src[row] * dinv[row]*w*dinv[col] ----------------
template<int F>
__global__ __launch_bounds__(256)
void edge_aggregate(const float* __restrict__ src, const int* __restrict__ rows,
                    const int* __restrict__ cols, const float* __restrict__ w,
                    const float* __restrict__ dinv, float* __restrict__ agg, int E)
{
    long idx = (long)blockIdx.x * 256 + threadIdx.x;
    int e = (int)(idx / F);
    int f = (int)(idx % F);
    if (e >= E) return;
    int r = rows[e], c = cols[e];
    float nw = dinv[r] * w[e] * dinv[c];
    atomicAdd(&agg[(size_t)c * F + f], src[(size_t)r * F + f] * nw);
}

// finalize: out = act(agg + self_loop + bias); optionally emit bf16
template<int F, int ACT, int TOBF>
__global__ __launch_bounds__(256)
void finalize_node(const float* __restrict__ agg, const float* __restrict__ src,
                   const float* __restrict__ dinv, const float* __restrict__ bias,
                   float* __restrict__ outf, __bf16* __restrict__ outb, int n)
{
    long idx = (long)blockIdx.x * 256 + threadIdx.x;
    if (idx >= (long)n * F) return;
    int i = (int)(idx / F), f = (int)(idx % F);
    float di = dinv[i];
    float v = agg[idx] + src[idx] * di * di + bias[f];
    if (ACT) v = fmaxf(v, 0.0f);
    if (TOBF) outb[idx] = (__bf16)v; else outf[idx] = v;
}

// ---------------- fused pair MLP: gather -> 128->64 relu -> 64->32 relu -> 32->1 ----------------
// 128 threads = 4 waves; each wave processes 16 pairs.
__global__ __launch_bounds__(128)
void pair_mlp_wmma(const __bf16* __restrict__ h2, const int* __restrict__ pidx,
                   const float* __restrict__ Wa, const float* __restrict__ ba,
                   const float* __restrict__ Wb, const float* __restrict__ bb,
                   const float* __restrict__ Wc, const float* __restrict__ bc,
                   float* __restrict__ logits)
{
    __shared__ __bf16 WaS[64][128];   // col-major [n][k], 16 KB
    __shared__ __bf16 WbS[32][64];    // col-major, 4 KB
    __shared__ float  baS[64], bbS[32], wcS[32];
    __shared__ __bf16 Ap[4][16][128]; // per-wave gathered inputs, 16 KB
    __shared__ __bf16 S1[4][16][64];  // per-wave stage-1 activations, 8 KB

    const int tid = threadIdx.x;
    for (int i = tid; i < 128 * 64; i += 128) { int k = i >> 6, n = i & 63; WaS[n][k] = (__bf16)Wa[i]; }
    for (int i = tid; i < 64 * 32;  i += 128) { int k = i >> 5, n = i & 31; WbS[n][k] = (__bf16)Wb[i]; }
    if (tid < 64) baS[tid] = ba[tid];
    if (tid < 32) { bbS[tid] = bb[tid]; wcS[tid] = Wc[tid]; }
    const float bcv = bc[0];
    __syncthreads();

    const int wave = tid >> 5, lane = tid & 31;
    const int ln16 = lane & 15, hi = lane >> 4;
    const long pbase = (long)blockIdx.x * 64 + wave * 16;

    // gather: lane L handles pair pbase+(L>>1), endpoint half=L&1 (64 bf16 = 128 B)
    {
        int p    = (int)pbase + (lane >> 1);
        int half = lane & 1;
        int node = pidx[(size_t)half * ES_PAIRS + p];
        const uint4* src = (const uint4*)(h2 + (size_t)node * 64);
        uint4* dst = (uint4*)&Ap[wave][lane >> 1][half * 64];
#pragma unroll
        for (int j = 0; j < 8; ++j) dst[j] = src[j];
    }
    __syncthreads();

    // stage 1: [16x128] @ [128x64]
    v8f acc1[4];
#pragma unroll
    for (int n = 0; n < 4; ++n) acc1[n] = (v8f){0,0,0,0,0,0,0,0};
#pragma unroll
    for (int ks = 0; ks < 4; ++ks) {
        const int k0 = ks * 32, kb0 = k0 + hi * 16, kb1 = k0 + hi * 16 + 8;
        v8bf lo = *(const v8bf*)&Ap[wave][ln16][kb0];
        v8bf hv = *(const v8bf*)&Ap[wave][ln16][kb1];
        v16bf a;
#pragma unroll
        for (int i = 0; i < 8; ++i) { a[i] = lo[i]; a[8 + i] = hv[i]; }
#pragma unroll
        for (int n = 0; n < 4; ++n) {
            v16bf b = *(const v16bf*)&WaS[n * 16 + ln16][k0 + hi * 16];
            acc1[n] = wmma_bf16(a, b, acc1[n]);
        }
    }
#pragma unroll
    for (int n = 0; n < 4; ++n)
#pragma unroll
        for (int r = 0; r < 8; ++r) {
            int m = hi * 8 + r, c = n * 16 + ln16;
            S1[wave][m][c] = (__bf16)fmaxf(acc1[n][r] + baS[c], 0.0f);
        }
    __syncthreads();

    // stage 2: [16x64] @ [64x32]
    v8f acc2[2];
#pragma unroll
    for (int n = 0; n < 2; ++n) acc2[n] = (v8f){0,0,0,0,0,0,0,0};
#pragma unroll
    for (int ks = 0; ks < 2; ++ks) {
        const int k0 = ks * 32, kb0 = k0 + hi * 16, kb1 = k0 + hi * 16 + 8;
        v8bf lo = *(const v8bf*)&S1[wave][ln16][kb0];
        v8bf hv = *(const v8bf*)&S1[wave][ln16][kb1];
        v16bf a;
#pragma unroll
        for (int i = 0; i < 8; ++i) { a[i] = lo[i]; a[8 + i] = hv[i]; }
#pragma unroll
        for (int n = 0; n < 2; ++n) {
            v16bf b = *(const v16bf*)&WbS[n * 16 + ln16][k0 + hi * 16];
            acc2[n] = wmma_bf16(a, b, acc2[n]);
        }
    }

    // stage 3: relu(s2) @ Wc + bc, reduced across the 16 columns per half-wave
    const float w0 = wcS[ln16], w1 = wcS[16 + ln16];
    float part[8];
#pragma unroll
    for (int r = 0; r < 8; ++r)
        part[r] = fmaxf(acc2[0][r] + bbS[ln16], 0.0f) * w0 +
                  fmaxf(acc2[1][r] + bbS[16 + ln16], 0.0f) * w1;
#pragma unroll
    for (int m = 1; m < 16; m <<= 1)
#pragma unroll
        for (int r = 0; r < 8; ++r) part[r] += __shfl_xor(part[r], m, 32);
    if (ln16 == 0) {
#pragma unroll
        for (int r = 0; r < 8; ++r)
            logits[pbase + hi * 8 + r] = part[r] + bcv;
    }
}

// ---------------- per-graph segment softmax ----------------
__device__ __forceinline__ void atomicMaxF(float* addr, float val) {
    if (val >= 0.0f) atomicMax((int*)addr, __float_as_int(val));
    else             atomicMin((unsigned int*)addr, (unsigned int)__float_as_int(val));
}

__global__ void seg_max_kernel(const float* __restrict__ lg, const int* __restrict__ g,
                               float* __restrict__ gmax, int n) {
    int i = blockIdx.x * blockDim.x + threadIdx.x;
    if (i < n) atomicMaxF(&gmax[g[i]], lg[i]);
}

__global__ void seg_expsum_kernel(float* __restrict__ lg, const int* __restrict__ g,
                                  const float* __restrict__ gmax, float* __restrict__ gsum, int n) {
    int i = blockIdx.x * blockDim.x + threadIdx.x;
    if (i < n) {
        float e = __expf(lg[i] - gmax[g[i]]);
        lg[i] = e;
        atomicAdd(&gsum[g[i]], e);
    }
}

__global__ void seg_norm_kernel(const float* __restrict__ lg, const int* __restrict__ g,
                                const float* __restrict__ gsum, float* __restrict__ out, int n) {
    int i = blockIdx.x * blockDim.x + threadIdx.x;
    if (i < n) out[i] = lg[i] / gsum[g[i]];
}

// ---------------- host orchestration ----------------
extern "C" void kernel_launch(void* const* d_in, const int* in_sizes, int n_in,
                              void* d_out, int out_size, void* d_ws, size_t ws_size,
                              hipStream_t stream)
{
    const float* x      = (const float*)d_in[0];
    const int*   eidx   = (const int*)  d_in[1];   // [2][E]
    const float* ew     = (const float*)d_in[2];
    const int*   pidx   = (const int*)  d_in[3];   // [2][ES]
    const int*   egraph = (const int*)  d_in[4];
    // d_in[5] = num_graphs scalar (compile-time constant here)
    const float* W_proj = (const float*)d_in[6];
    const float* b_proj = (const float*)d_in[7];
    const float* W1     = (const float*)d_in[8];
    const float* b1     = (const float*)d_in[9];
    const float* W2     = (const float*)d_in[10];
    const float* b2     = (const float*)d_in[11];
    const float* Wa     = (const float*)d_in[12];
    const float* ba     = (const float*)d_in[13];
    const float* Wb     = (const float*)d_in[14];
    const float* bb     = (const float*)d_in[15];
    const float* Wc     = (const float*)d_in[16];
    const float* bc     = (const float*)d_in[17];

    char* ws = (char*)d_ws;
    auto carve = [&](size_t bytes) -> char* {
        char* p = ws;
        ws += (bytes + 255) & ~(size_t)255;
        return p;
    };
    float*  dinv  = (float*) carve((size_t)N_NODES * 4);          // deg -> dinv
    float*  h     = (float*) carve((size_t)N_NODES * 128 * 4);    // h0, later h1
    float*  t     = (float*) carve((size_t)N_NODES * 128 * 4);    // hw1, later hw2
    float*  agg   = (float*) carve((size_t)N_NODES * 128 * 4);    // aggregation target
    __bf16* h2    = (__bf16*)carve((size_t)N_NODES * 64 * 2);     // layer-2 output, bf16
    float*  logit = (float*) carve((size_t)ES_PAIRS * 4);
    float*  gmax  = (float*) carve((size_t)G_GRAPHS * 4);
    float*  gsum  = (float*) carve((size_t)G_GRAPHS * 4);

    const int*   rows = eidx;             // edge_index[0]
    const int*   cols = eidx + E_EDGES;   // edge_index[1]

    const int TB = 256;
    // 1) degree & normalization
    fill_kernel<<<(N_NODES + TB - 1) / TB, TB, 0, stream>>>(dinv, 0.0f, N_NODES);
    degree_kernel<<<(E_EDGES + TB - 1) / TB, TB, 0, stream>>>(cols, ew, dinv, E_EDGES);
    dinv_kernel<<<(N_NODES + TB - 1) / TB, TB, 0, stream>>>(dinv, N_NODES);

    const int gemm_blocks = (N_NODES + 63) / 64;
    // 2) h0 = x @ W_proj + b_proj
    gemm_wmma<64, 128, 0><<<gemm_blocks, 128, 0, stream>>>(x, W_proj, b_proj, h, N_NODES);
    // 3) hw1 = h0 @ W1
    gemm_wmma<128, 128, 0><<<gemm_blocks, 128, 0, stream>>>(h, W1, nullptr, t, N_NODES);
    // 4) layer-1 aggregation + self loops + bias + relu -> h1 (in-place over h)
    fill_kernel<<<((long)N_NODES * 128 + TB - 1) / TB, TB, 0, stream>>>(agg, 0.0f, (long)N_NODES * 128);
    edge_aggregate<128><<<((long)E_EDGES * 128 + TB - 1) / TB, TB, 0, stream>>>(t, rows, cols, ew, dinv, agg, E_EDGES);
    finalize_node<128, 1, 0><<<((long)N_NODES * 128 + TB - 1) / TB, TB, 0, stream>>>(agg, t, dinv, b1, h, nullptr, N_NODES);
    // 5) hw2 = h1 @ W2
    gemm_wmma<128, 64, 0><<<gemm_blocks, 128, 0, stream>>>(h, W2, nullptr, t, N_NODES);
    // 6) layer-2 aggregation + self loops + bias -> h2 (bf16)
    fill_kernel<<<((long)N_NODES * 64 + TB - 1) / TB, TB, 0, stream>>>(agg, 0.0f, (long)N_NODES * 64);
    edge_aggregate<64><<<((long)E_EDGES * 64 + TB - 1) / TB, TB, 0, stream>>>(t, rows, cols, ew, dinv, agg, E_EDGES);
    finalize_node<64, 0, 1><<<((long)N_NODES * 64 + TB - 1) / TB, TB, 0, stream>>>(agg, t, dinv, b2, nullptr, h2, N_NODES);
    // 7) fused pair MLP -> logits
    pair_mlp_wmma<<<ES_PAIRS / 64, 128, 0, stream>>>(h2, pidx, Wa, ba, Wb, bb, Wc, bc, logit);
    // 8) per-graph segment softmax
    fill_kernel<<<1, TB, 0, stream>>>(gmax, -INFINITY, G_GRAPHS);
    fill_kernel<<<1, TB, 0, stream>>>(gsum, 0.0f, G_GRAPHS);
    seg_max_kernel<<<(ES_PAIRS + TB - 1) / TB, TB, 0, stream>>>(logit, egraph, gmax, ES_PAIRS);
    seg_expsum_kernel<<<(ES_PAIRS + TB - 1) / TB, TB, 0, stream>>>(logit, egraph, gmax, gsum, ES_PAIRS);
    seg_norm_kernel<<<(ES_PAIRS + TB - 1) / TB, TB, 0, stream>>>(logit, egraph, gsum, (float*)d_out, ES_PAIRS);
}
